// GNN_59700045414568
// MI455X (gfx1250) — matrix operand order repaired and checked by reading, not compiled, not run on hardware
//
#include <hip/hip_runtime.h>

typedef __bf16 bf16;
typedef __attribute__((ext_vector_type(16))) __bf16 v16bf;
typedef __attribute__((ext_vector_type(8)))  __bf16 v8bf;
typedef __attribute__((ext_vector_type(8)))  float   v8f;
typedef __attribute__((ext_vector_type(4)))  float   v4f;
typedef __attribute__((ext_vector_type(4)))  unsigned int v4u;
typedef __attribute__((ext_vector_type(8)))  int     v8i;
typedef __attribute__((ext_vector_type(4)))  int     v4i;

namespace {
constexpr int kE   = 320000;   // edges
constexpr int kN   = 20000;    // nodes
constexpr int kFN  = 64;       // node feat
constexpr int kFE  = 16;       // edge feat
constexpr int kHID = 300;
constexpr int kKP  = 320;      // padded hidden pitch (multiple of 32)
constexpr int kG   = 128;      // graphs
constexpr int kKI  = 96;       // padded K for init layer (64+16 -> 96)
constexpr int kKE  = 384;      // padded K for e2n (64+300 -> 384)
constexpr int kBP  = 40;       // LDS row pitch (32 + 8 pad) to spread banks
}

__device__ __forceinline__ float bf2f(bf16 v) { return (float)v; }

__device__ __forceinline__ unsigned lds_addr(const void* p) {
  // generic pointers to LDS carry the DS byte offset in the low 32 bits
  return (unsigned)(unsigned long long)p;
}

__device__ __forceinline__ void async_ld_b128(unsigned lds, const bf16* gaddr) {
  asm volatile("global_load_async_to_lds_b128 %0, %1, off"
               :: "v"(lds), "v"(gaddr) : "memory");
}

__device__ __forceinline__ void wait_async0() {
  asm volatile("s_wait_asynccnt 0x0" ::: "memory");
}

// TDM: 2D tile load Global->LDS. tile0 elems/row (2B each), tile1 rows,
// stride0 = global row stride in elements. LDS rows padded +16B per 64B row
// (pad_interval=3 -> every 16 DWORDs, pad_amount=3 -> 4 DWORDs) == kBP pitch.
__device__ __forceinline__ void tdm_load_2d(unsigned lds, const bf16* gaddr,
                                            int tile0, int tile1, long stride0) {
  unsigned long long ga = (unsigned long long)gaddr;
  unsigned td0 = (unsigned)tile0, td1 = (unsigned)tile1;
  v4u g0 = {0u, 0u, 0u, 0u};
  g0[0] = 1u;                                      // count=1, user descriptor
  g0[1] = lds;                                     // lds_addr (bytes)
  g0[2] = (unsigned)ga;                            // global_addr[31:0]
  g0[3] = (unsigned)((ga >> 32) & 0x01FFFFFFull)   // global_addr[56:32]
        | (2u << 30);                              // type=2 ("image")
  v8i g1 = {0, 0, 0, 0, 0, 0, 0, 0};
  g1[0] = (int)((1u << 16)                         // data_size = 2 bytes
        | (1u << 20) | (3u << 22) | (3u << 25));   // pad_enable, every 16 DW, +4 DW
  g1[1] = (int)((td0 & 0xFFFFu) << 16);            // tensor_dim0[15:0]
  g1[2] = (int)(((td0 >> 16) & 0xFFFFu) | ((td1 & 0xFFFFu) << 16));   // td0 hi | td1 lo
  g1[3] = (int)(((td1 >> 16) & 0xFFFFu) | ((td0 & 0xFFFFu) << 16));   // td1 hi | tile_dim0
  g1[4] = (int)(td1 & 0xFFFFu);                    // tile_dim1 (tile_dim2=0)
  g1[5] = (int)((unsigned long long)stride0 & 0xFFFFFFFFull);          // dim0 stride lo
  g1[6] = (int)(((unsigned long long)stride0 >> 32) & 0xFFFFull);      // dim0 stride hi
  v4i gz = {0, 0, 0, 0};
#if __clang_major__ >= 23
  v8i gz8 = {0, 0, 0, 0, 0, 0, 0, 0};
  __builtin_amdgcn_tensor_load_to_lds(g0, g1, gz, gz, gz8, 0);
#else
  __builtin_amdgcn_tensor_load_to_lds(g0, g1, gz, gz, 0);
#endif
}

// ---------------- weight / bias prep ----------------
// Wt[n][k] = W[k][n] (bf16, zero padded): B-fragment friendly (n-major, contiguous K)
__global__ void k_prep_w(const float* __restrict__ W, bf16* __restrict__ Wt,
                         int Kreal, int Nreal, int Kpad, int Npad) {
  int idx = blockIdx.x * blockDim.x + threadIdx.x;
  if (idx >= Kpad * Npad) return;
  int n = idx / Kpad;
  int k = idx - n * Kpad;
  float v = (k < Kreal && n < Nreal) ? W[(long)k * Nreal + n] : 0.0f;
  Wt[idx] = (bf16)v;
}

__global__ void k_prep_b(const float* __restrict__ b, float* __restrict__ bp, int Nreal) {
  int n = blockIdx.x * blockDim.x + threadIdx.x;
  if (n < kKP) bp[n] = (n < Nreal) ? b[n] : 0.0f;
}

__global__ void k_zero(v4f* __restrict__ p, long n4) {
  long i = (long)blockIdx.x * blockDim.x + threadIdx.x;
  if (i < n4) {
    v4f z = {0.f, 0.f, 0.f, 0.f};
    p[i] = z;
  }
}

// ---------------- segment sum: a[col[e]][k] += h[e][k] ----------------
__global__ void k_seg_sum(const bf16* __restrict__ h, const int* __restrict__ col,
                          float* __restrict__ a) {
  long idx = (long)blockIdx.x * blockDim.x + threadIdx.x;
  constexpr long tot = (long)kE * (kKP / 8);
  if (idx >= tot) return;
  int e  = (int)(idx / (kKP / 8));
  int k8 = (int)(idx % (kKP / 8)) * 8;
  int c  = col[e];
  v8bf hv = *(const v8bf*)(h + (long)e * kKP + k8);
  float* ap = a + (long)c * kKP + k8;
#pragma unroll
  for (int i = 0; i < 8; ++i) unsafeAtomicAdd(ap + i, bf2f(hv[i]));
}

// ---------------- WMMA fragment loaders (CDNA5 wave32 layouts) ----------------
// A 16x32 bf16: lane m=lane%16, kbase=(lane<16)?0:8; V0-3 = K[kbase..+7], V4-7 = K[kbase+16..+23]
__device__ __forceinline__ v16bf frag_a(const bf16* __restrict__ As, int lane, int msub) {
  const bf16* r = As + (msub * 16 + (lane & 15)) * kBP + ((lane & 16) ? 8 : 0);
  v16bf f;
  ((v8bf*)&f)[0] = *(const v8bf*)(r);
  ((v8bf*)&f)[1] = *(const v8bf*)(r + 16);
  return f;
}
// B 32x16 bf16 (n-major, 32 K contiguous per n): lane n=lane%16, kbase=(lane<16)?0:8; V0-3 = K[kbase..+7], V4-7 = K[kbase+16..+23]
__device__ __forceinline__ v16bf frag_b(const bf16* __restrict__ Bs, int lane, int ntile) {
  const bf16* r = Bs + (ntile * 16 + (lane & 15)) * kBP + ((lane & 16) ? 8 : 0);
  v16bf f;
  ((v8bf*)&f)[0] = *(const v8bf*)(r);
  ((v8bf*)&f)[1] = *(const v8bf*)(r + 16);
  return f;
}

// ---------------- unified pipelined GEMM ----------------
// MODE 0 (INIT): h0out = relu(cat(x[row], eattr) @ W + b)         [async-LDS staging]
// MODE 1 (CONV): hout  = relu((a[row] - h_in[e^1]) @ W + b + h0)  [TDM staging]
// MODE 2 (E2N):  pooled[batch[v]] += relu(cat(x, s) @ W + b)      [async-LDS staging]
struct APre { v4f lo, hi; v8bf hv; };

template <int MODE>
__device__ __forceinline__ APre prefA(int kt, const float* __restrict__ f32a,
                                      const float* __restrict__ f32b,
                                      const bf16* __restrict__ hrev,
                                      long rowA, long mGlob, bool valid, int kc) {
  APre p;
  v4f z = {0.f, 0.f, 0.f, 0.f};
  p.lo = z; p.hi = z;
  if (MODE == 0) {
    if (kt < 2) {                       // x region (k < 64)
      const float* xp = f32a + rowA * kFN + kt * 32 + kc;
      p.lo = *(const v4f*)xp; p.hi = *(const v4f*)(xp + 4);
    } else if (kc < 16) {               // edge_attr region (64..79), rest zero pad
      const float* ep = f32b + mGlob * kFE + kc;
      p.lo = *(const v4f*)ep; p.hi = *(const v4f*)(ep + 4);
    }
  } else if (MODE == 1) {
    const float* ap = f32a + rowA * kKP + kt * 32 + kc;
    p.lo = *(const v4f*)ap; p.hi = *(const v4f*)(ap + 4);
    p.hv = *(const v8bf*)(hrev + (mGlob ^ 1) * kKP + kt * 32 + kc);
  } else {
    if (valid) {
      const float* sp = (kt < 2) ? (f32a + mGlob * kFN + kt * 32 + kc)
                                 : (f32b + mGlob * kKP + (kt * 32 + kc - 64));
      p.lo = *(const v4f*)sp; p.hi = *(const v4f*)(sp + 4);
    }
  }
  return p;
}

template <int MODE>
__device__ __forceinline__ void commitA(bf16* __restrict__ dst, const APre& p) {
  v8bf o;
#pragma unroll
  for (int i = 0; i < 8; ++i) {
    float v = (i < 4) ? p.lo[i] : p.hi[i - 4];
    if (MODE == 1) v -= bf2f(p.hv[i]);
    o[i] = (bf16)v;
  }
  *(v8bf*)dst = o;
}

template <int MODE, int KSTEPS>
__global__ __launch_bounds__(256)
void k_gemm(const float* __restrict__ f32a, const float* __restrict__ f32b,
            const bf16* __restrict__ hrev, const bf16* __restrict__ h0,
            const int* __restrict__ idx, const bf16* __restrict__ Wt, int wpitch,
            const float* __restrict__ bias, bf16* __restrict__ hout,
            float* __restrict__ pooled) {
  __shared__ __align__(16) bf16 As[2][64 * kBP];
  __shared__ __align__(16) bf16 Bs[2][kKP * kBP];

  const int  tid   = threadIdx.x;
  const long mBase = (long)blockIdx.x * 64;
  const int  lane  = tid & 31, wv = tid >> 5;
  const int  msub  = wv & 3, nhalf = wv >> 2;
  const int  mA    = tid >> 2;          // 0..63: row this thread stages
  const int  kc    = (tid & 3) * 8;     // 0,8,16,24: K chunk this thread stages
  const long mGlob = mBase + mA;
  const bool mValid = (MODE == 2) ? (mGlob < kN) : true;
  long rowA = 0;
  if (MODE == 0 || MODE == 1) rowA = idx[mGlob];

  v8f acc[10];
  const v8f z8 = {0.f,0.f,0.f,0.f,0.f,0.f,0.f,0.f};
#pragma unroll
  for (int t = 0; t < 10; ++t) acc[t] = z8;

  // stage B slice kt into buffer buf
  auto stageB = [&](int kt, int buf) {
    if (MODE == 1) {
      if (wv == 0)            // one TDM descriptor per workgroup (wave-uniform)
        tdm_load_2d(lds_addr(&Bs[buf][0]), Wt + kt * 32, 32, kKP, (long)wpitch);
    } else {
#pragma unroll
      for (int c = 0; c < 5; ++c) {
        int chunk = tid + c * 256;      // 0..1279 : 320 rows x 4 chunks
        int n  = chunk >> 2;
        int cc = (chunk & 3) * 8;
        async_ld_b128(lds_addr(&Bs[buf][n * kBP + cc]),
                      Wt + (long)n * wpitch + kt * 32 + cc);
      }
    }
  };
  auto waitB = [&]() {
    if (MODE == 1) {
      if (wv == 0) __builtin_amdgcn_s_wait_tensorcnt((short)0);
    } else {
      wait_async0();
    }
  };

  // ---- prologue: stage slice 0 ----
  stageB(0, 0);
  {
    APre p0 = prefA<MODE>(0, f32a, f32b, hrev, rowA, mGlob, mValid, kc);
    commitA<MODE>(&As[0][mA * kBP + kc], p0);
  }
  waitB();
  __syncthreads();

  // ---- pipelined K loop ----
#pragma unroll 1
  for (int kt = 0; kt < KSTEPS; ++kt) {
    const int cur = kt & 1, nxt = cur ^ 1;
    APre pn;
    if (kt + 1 < KSTEPS) {
      stageB(kt + 1, nxt);                                   // TDM/async HBM->LDS (B)
      pn = prefA<MODE>(kt + 1, f32a, f32b, hrev, rowA, mGlob, mValid, kc);  // A into regs
    }
    v16bf af = frag_a(&As[cur][0], lane, msub);
    // register double-buffer of B fragments: load t+1 before WMMA t so the
    // ds_load latency hides under the matrix op instead of s_wait_dscnt 0 stalls
    v16bf bcur = frag_b(&Bs[cur][0], lane, nhalf * 10);
#pragma unroll
    for (int t = 0; t < 10; ++t) {
      v16bf bnxt = bcur;
      if (t + 1 < 10) bnxt = frag_b(&Bs[cur][0], lane, nhalf * 10 + t + 1);
      acc[t] = __builtin_amdgcn_wmma_f32_16x16x32_bf16(false, af, false, bcur,
                                                       (short)0, acc[t], false, false);
      bcur = bnxt;
    }
    if (kt + 1 < KSTEPS) commitA<MODE>(&As[nxt][mA * kBP + kc], pn);
    waitB();
    __syncthreads();
  }

  // ---- epilogue ----
#pragma unroll
  for (int t = 0; t < 10; ++t) {
    const int n  = (nhalf * 10 + t) * 16 + (lane & 15);
    const float bn = bias[n];
#pragma unroll
    for (int j = 0; j < 8; ++j) {
      const int  ml = msub * 16 + j + ((lane & 16) ? 8 : 0);
      const long m  = mBase + ml;
      if (MODE == 2) {
        if (m < kN) {
          float hn = acc[t][j] + bn;
          hn = hn > 0.f ? hn : 0.f;
          unsafeAtomicAdd(pooled + (long)idx[m] * kKP + n, hn);
        }
      } else {
        float v = acc[t][j] + bn;
        if (MODE == 1) v += bf2f(h0[m * kKP + n]);
        v = v > 0.f ? v : 0.f;
        hout[m * kKP + n] = (bf16)v;
      }
    }
  }
}

// ---------------- final FFN: out[g] = pooled[g] . W_ffn + b ----------------
__global__ void k_ffn(const float* __restrict__ pooled, const float* __restrict__ Wf,
                      const float* __restrict__ bf_, float* __restrict__ out) {
  int g = blockIdx.x * blockDim.x + threadIdx.x;
  if (g >= kG) return;
  float sum = 0.f;
  for (int n = 0; n < kHID; ++n) sum += pooled[(long)g * kKP + n] * Wf[n];
  out[g] = sum + bf_[0];
}

// ---------------- host-side orchestration ----------------
extern "C" void kernel_launch(void* const* d_in, const int* in_sizes, int n_in,
                              void* d_out, int out_size, void* d_ws, size_t ws_size,
                              hipStream_t stream) {
  (void)in_sizes; (void)n_in; (void)out_size; (void)ws_size;
  const float* x       = (const float*)d_in[0];
  const float* eattr   = (const float*)d_in[1];
  const int*   eidx    = (const int*)d_in[2];
  const int*   batch   = (const int*)d_in[3];
  const float* W_init  = (const float*)d_in[4];
  const float* b_init  = (const float*)d_in[5];
  const float* W_convs = (const float*)d_in[6];
  const float* b_convs = (const float*)d_in[7];
  const float* W_e2n   = (const float*)d_in[8];
  const float* b_e2n   = (const float*)d_in[9];
  const float* W_ffn   = (const float*)d_in[10];
  const float* b_ffn   = (const float*)d_in[11];
  float* out = (float*)d_out;

  const int* row = eidx;
  const int* col = eidx + kE;

  char* wsp = (char*)d_ws;
  auto alloc = [&](size_t bytes) -> char* {
    char* p = wsp;
    wsp += (bytes + 255) & ~(size_t)255;
    return p;
  };
  bf16*  h0     = (bf16*) alloc((size_t)kE * kKP * 2);
  bf16*  hA     = (bf16*) alloc((size_t)kE * kKP * 2);
  bf16*  hB     = (bf16*) alloc((size_t)kE * kKP * 2);
  float* a      = (float*)alloc((size_t)kN * kKP * 4);
  float* pooled = (float*)alloc((size_t)kG * kKP * 4);
  bf16*  WtI    = (bf16*) alloc((size_t)kKP * kKI * 2);
  bf16*  WtC    = (bf16*) alloc((size_t)3 * kKP * kKP * 2);
  bf16*  WtE    = (bf16*) alloc((size_t)kKP * kKE * 2);
  float* bI     = (float*)alloc((size_t)kKP * 4);
  float* bC     = (float*)alloc((size_t)3 * kKP * 4);
  float* bE     = (float*)alloc((size_t)kKP * 4);

  // weight / bias prep (transpose + pad + bf16)
  k_prep_w<<<(kKP * kKI + 255) / 256, 256, 0, stream>>>(W_init, WtI, kFN + kFE, kHID, kKI, kKP);
  for (int l = 0; l < 3; ++l)
    k_prep_w<<<(kKP * kKP + 255) / 256, 256, 0, stream>>>(
        W_convs + (size_t)l * kHID * kHID, WtC + (size_t)l * kKP * kKP, kHID, kHID, kKP, kKP);
  k_prep_w<<<(kKP * kKE + 255) / 256, 256, 0, stream>>>(W_e2n, WtE, kFN + kHID, kHID, kKE, kKP);
  k_prep_b<<<2, 256, 0, stream>>>(b_init, bI, kHID);
  for (int l = 0; l < 3; ++l)
    k_prep_b<<<2, 256, 0, stream>>>(b_convs + (size_t)l * kHID, bC + (size_t)l * kKP, kHID);
  k_prep_b<<<2, 256, 0, stream>>>(b_e2n, bE, kHID);

  // h0 = relu(cat(x[row], edge_attr) @ W_init + b)
  k_gemm<0, kKI / 32><<<kE / 64, 256, 0, stream>>>(
      x, eattr, nullptr, nullptr, row, WtI, kKI, bI, h0, nullptr);

  bf16* hs[4] = {h0, hA, hB, hA};  // ping-pong: h0->hA->hB->hA
  const long a4   = (long)kN * kKP / 4;
  const long segT = (long)kE * (kKP / 8);
  for (int l = 0; l < 3; ++l) {
    k_zero<<<(a4 + 255) / 256, 256, 0, stream>>>((v4f*)a, a4);
    k_seg_sum<<<(segT + 255) / 256, 256, 0, stream>>>(hs[l], col, a);
    k_gemm<1, kKP / 32><<<kE / 64, 256, 0, stream>>>(
        a, nullptr, hs[l], h0, row, WtC + (size_t)l * kKP * kKP, kKP,
        bC + (size_t)l * kKP, hs[l + 1], nullptr);
  }

  // s = segment_sum(h_final, col)  (reuse a)
  k_zero<<<(a4 + 255) / 256, 256, 0, stream>>>((v4f*)a, a4);
  k_seg_sum<<<(segT + 255) / 256, 256, 0, stream>>>(hs[3], col, a);

  // pooled[g] += relu(cat(x, s) @ W_e2n + b)
  const long p4 = (long)kG * kKP / 4;
  k_zero<<<(p4 + 255) / 256, 256, 0, stream>>>((v4f*)pooled, p4);
  k_gemm<2, kKE / 32><<<(kN + 63) / 64, 256, 0, stream>>>(
      x, a, nullptr, nullptr, batch, WtE, kKE, bE, nullptr, pooled);

  // out = pooled @ W_ffn + b_ffn
  k_ffn<<<1, kG, 0, stream>>>(pooled, W_ffn, b_ffn, out);
}
